// TimeTRMBlock_77446850282055
// MI455X (gfx1250) — compile-verified, matrix operand order checked by reading, not run on hardware
//
#include <hip/hip_runtime.h>

// ---------------------------------------------------------------------------
// TimeTRMBlock for MI455X (gfx1250, wave32, WMMA f32<-f16 16x16x32, TDM)
//   B=1, L=512, D=128, H=4, hd=32, NB=2
// ---------------------------------------------------------------------------

#define LSEQ 512
#define DIM  128
#define NHEAD 4
#define HDIM 32
#define NBLK 2
#define NEGV (-4294967295.0f)

#ifndef __has_builtin
#define __has_builtin(x) 0
#endif

#if __has_builtin(__builtin_amdgcn_tensor_load_to_lds) && \
    __has_builtin(__builtin_amdgcn_s_wait_tensorcnt)
#define USE_TDM 1
#else
#define USE_TDM 0
#endif

#if __has_builtin(__builtin_amdgcn_update_dpp)
#define USE_DPP 1
#else
#define USE_DPP 0
#endif

typedef __attribute__((ext_vector_type(16))) _Float16 v16h;
typedef __attribute__((ext_vector_type(8)))  float    v8f;
typedef __attribute__((ext_vector_type(4)))  unsigned v4u;
typedef __attribute__((ext_vector_type(8)))  int      v8i;
typedef __attribute__((ext_vector_type(4)))  int      v4i;

__device__ __forceinline__ v8f wmma16x16x32(v16h a, v16h b, v8f c) {
  // D = A(16x32 f16) * B(32x16 f16) + C(16x16 f32)
  return __builtin_amdgcn_wmma_f32_16x16x32_f16(
      /*neg_a=*/false, a, /*neg_b=*/false, b,
      /*c_mod=*/(short)0, c, /*reuse_a=*/false, /*reuse_b=*/false);
}

// Sum across the 16 lanes of a DPP row; every lane ends with the row total.
// Steps: lane^1, lane^2 (quad_perm), lane^7 (row_half_mirror: exact xor4 once
// quads are uniform), lane^15 (row_mirror: exact xor8 once 8-groups uniform).
__device__ __forceinline__ float row16_sum(float x) {
#if USE_DPP
  int v;
  v = __builtin_amdgcn_update_dpp(0, __float_as_int(x), 0xB1, 0xF, 0xF, true);
  x += __int_as_float(v);
  v = __builtin_amdgcn_update_dpp(0, __float_as_int(x), 0x4E, 0xF, 0xF, true);
  x += __int_as_float(v);
  v = __builtin_amdgcn_update_dpp(0, __float_as_int(x), 0x141, 0xF, 0xF, true);
  x += __int_as_float(v);
  v = __builtin_amdgcn_update_dpp(0, __float_as_int(x), 0x140, 0xF, 0xF, true);
  x += __int_as_float(v);
  return x;
#else
#pragma unroll
  for (int m = 8; m >= 1; m >>= 1) x += __shfl_xor(x, m, 32);
  return x;
#endif
}

#if USE_TDM
// TDM: DMA one contiguous run of `n` f32 elements from global to LDS.
// D# per CDNA5 ISA 8.3/8.4; 1-D copy (tile_dim1=1) so strides are inert.
__device__ __forceinline__ void tdm_load_f32(unsigned lds_off,
                                             const float* gsrc,
                                             unsigned n_tile,
                                             unsigned n_tensor) {
  unsigned long ga = (unsigned long)(const void*)gsrc;
  v4u g0 = { 1u,                                   // count = 1 (valid)
             lds_off,                              // lds_addr (bytes)
             (unsigned)(ga & 0xffffffffu),         // global_addr[31:0]
             (unsigned)((ga >> 32) & 0x1ffffffu)   // global_addr[56:32]
                 | (2u << 30) };                   // type = 2 ("image")
  v8i g1 = { (int)(2u << 16),                              // data_size = 4B
             (int)((n_tensor & 0xffffu) << 16),            // tensor_dim0 lo16
             (int)(((n_tensor >> 16) & 0xffffu) | (1u << 16)), // dim0 hi | dim1=1
             (int)(n_tile << 16),                          // dim1 hi=0 | tile_dim0
             1,                                            // tile_dim1=1, tile_dim2=0
             (int)n_tensor,                                // dim0_stride lo32
             0, 0 };                                       // stride hi bits
  v4i gz = { 0, 0, 0, 0 };
#if __clang_major__ >= 23
  v8i gz8 = { 0, 0, 0, 0, 0, 0, 0, 0 };
  __builtin_amdgcn_tensor_load_to_lds(g0, g1, gz, gz, gz8, 0);
#else
  __builtin_amdgcn_tensor_load_to_lds(g0, g1, gz, gz, 0);
#endif
}
#endif

// ---------------------------------------------------------------------------
// Kernel 1: time bias
//   logtw[i][q][k] = log(sigmoid(relu(tmK[q,k,:] @ Wt[i] + bt[i]) @ Wtp[i] + btp[i]))
// Wt B-fragments register-resident per wave; 16x128 f32 A tiles DMA'd into a
// double-buffered LDS stage by the Tensor Data Mover (s_wait_tensorcnt), then
// converted once to f16 for the WMMA fragment reads.
// ---------------------------------------------------------------------------
__global__ __launch_bounds__(256)
void time_bias_kernel(const float* __restrict__ tmK,   // [L*L, D]
                      const float* __restrict__ Wt,    // [NB, D, D]
                      const float* __restrict__ bt,    // [NB, D]
                      const float* __restrict__ Wtp,   // [NB, D]
                      const float* __restrict__ btp,   // [NB]
                      float* __restrict__ logtw)       // [NB, L*L]
{
  const int blk   = blockIdx.y;
  const int tid   = threadIdx.x;
  const int lane  = tid & 31;
  const int wave  = tid >> 5;
  const int ln15  = lane & 15;
  const int halfk = (lane >> 4) * 8;
  const int col   = wave * 16 + ln15;

  __shared__ _Float16 a16[16 * DIM];     // f16 A tile for fragment reads
  __shared__ float    p_lds[16];
#if USE_TDM
  __shared__ float    stage[2][16 * DIM];  // TDM double buffer (2 x 8KB)
#endif

  // Register-resident B fragments for this wave's 16-column strip of Wt.
  const float* W = Wt + blk * DIM * DIM;
  v16h bfrag[4];
#pragma unroll
  for (int kc = 0; kc < 4; ++kc) {
#pragma unroll
    for (int i = 0; i < 8; ++i) {
      bfrag[kc][i]     = (_Float16)W[(kc * 32 + halfk + i) * DIM + col];
      bfrag[kc][i + 8] = (_Float16)W[(kc * 32 + halfk + 16 + i) * DIM + col];
    }
  }
  const float btc  = bt[blk * DIM + col];
  const float wtpc = Wtp[blk * DIM + col];
  const float btpv = btp[blk];

  const int tilesPerBlock = (LSEQ * LSEQ / 16) / gridDim.x;
  const int tile0 = blockIdx.x * tilesPerBlock;
  const unsigned n_tensor = (unsigned)LSEQ * LSEQ * DIM;

#if USE_TDM
  if (wave == 0)
    tdm_load_f32((unsigned)(unsigned long)(void*)&stage[0][0],
                 tmK + (long)tile0 * 16 * DIM, 16 * DIM, n_tensor);
#endif

  for (int it = 0; it < tilesPerBlock; ++it) {
    const int tile = tile0 + it;

#if USE_TDM
    const int cur = it & 1;
    if (wave == 0) __builtin_amdgcn_s_wait_tensorcnt(0);
    __syncthreads();   // tile data visible to all waves; prior compute retired
    if (wave == 0 && it + 1 < tilesPerBlock)
      tdm_load_f32((unsigned)(unsigned long)(void*)&stage[cur ^ 1][0],
                   tmK + (long)(tile + 1) * 16 * DIM, 16 * DIM, n_tensor);
    // convert staged f32 tile -> f16 (8 elems per thread, packed cvts)
#pragma unroll
    for (int i = 0; i < 8; ++i)
      a16[tid * 8 + i] = (_Float16)stage[cur][tid * 8 + i];
#else
    const float* asrc = tmK + (long)tile * 16 * DIM;
#pragma unroll
    for (int i = 0; i < 8; ++i)
      a16[tid * 8 + i] = (_Float16)asrc[tid * 8 + i];
#endif
    if (tid < 16) p_lds[tid] = 0.0f;
    __syncthreads();

    // GEMM: 16 rows x 16 cols (this wave) over K=128
    v8f c = {};
#pragma unroll
    for (int kc = 0; kc < 4; ++kc) {
      v16h a;
      const _Float16* ap = &a16[ln15 * DIM + kc * 32 + halfk];
#pragma unroll
      for (int i = 0; i < 8; ++i) { a[i] = ap[i]; a[i + 8] = ap[16 + i]; }
      c = wmma16x16x32(a, bfrag[kc], c);
    }

    // relu(+bias), dot with Wtp column; DPP row reduction over 16 lanes
    float pr[8];
#pragma unroll
    for (int r = 0; r < 8; ++r) {
      float t = c[r] + btc;
      t = t > 0.0f ? t : 0.0f;
      pr[r] = row16_sum(t * wtpc);
    }
    if (ln15 == 0) {
#pragma unroll
      for (int r = 0; r < 8; ++r)
        atomicAdd(&p_lds[halfk + r], pr[r]);   // ds_add_f32 across 8 waves
    }
    __syncthreads();

    if (tid < 16) {
      float x = p_lds[tid] + btpv;
      // log(sigmoid(x)) = -log1p(exp(-x))
      logtw[(long)blk * LSEQ * LSEQ + (long)tile * 16 + tid] =
          -log1pf(__expf(-x));
    }
  }
}

// ---------------------------------------------------------------------------
// Kernel 2: generic C = A[512,128] @ W[128,128] + bias (+optional relu)
// ---------------------------------------------------------------------------
__global__ __launch_bounds__(256)
void gemm_tile16(const float* __restrict__ A,    // [L, D]
                 const float* __restrict__ W,    // [D, D] row-major (k,n)
                 const float* __restrict__ bias, // [D]
                 float* __restrict__ out,        // [L, D]
                 int do_relu)
{
  const int tid   = threadIdx.x;
  const int lane  = tid & 31;
  const int wave  = tid >> 5;
  const int ln15  = lane & 15;
  const int halfk = (lane >> 4) * 8;
  const int col   = wave * 16 + ln15;
  const int tile  = blockIdx.x;

  __shared__ _Float16 a_lds[16 * DIM];

  v16h bfrag[4];
#pragma unroll
  for (int kc = 0; kc < 4; ++kc) {
#pragma unroll
    for (int i = 0; i < 8; ++i) {
      bfrag[kc][i]     = (_Float16)W[(kc * 32 + halfk + i) * DIM + col];
      bfrag[kc][i + 8] = (_Float16)W[(kc * 32 + halfk + 16 + i) * DIM + col];
    }
  }

  const float* asrc = A + tile * 16 * DIM;
#pragma unroll
  for (int i = 0; i < 8; ++i)
    a_lds[tid * 8 + i] = (_Float16)asrc[tid * 8 + i];
  __syncthreads();

  v8f c = {};
#pragma unroll
  for (int kc = 0; kc < 4; ++kc) {
    v16h a;
    const _Float16* ap = &a_lds[ln15 * DIM + kc * 32 + halfk];
#pragma unroll
    for (int i = 0; i < 8; ++i) { a[i] = ap[i]; a[i + 8] = ap[16 + i]; }
    c = wmma16x16x32(a, bfrag[kc], c);
  }

  const float bcol = bias[col];
#pragma unroll
  for (int r = 0; r < 8; ++r) {
    float v = c[r] + bcol;
    if (do_relu) v = v > 0.0f ? v : 0.0f;
    out[(tile * 16 + halfk + r) * DIM + col] = v;
  }
}

// ---------------------------------------------------------------------------
// Kernel 3: layernorm over D=128, one wave per row (4 rows / 128-thread block)
// ---------------------------------------------------------------------------
__global__ __launch_bounds__(128)
void ln_kernel(const float* __restrict__ x, const float* __restrict__ g,
               const float* __restrict__ b, float* __restrict__ out)
{
  const int lane = threadIdx.x & 31;
  const int wave = threadIdx.x >> 5;
  const int row  = blockIdx.x * 4 + wave;

  const float4 v = ((const float4*)(x + row * DIM))[lane];
  float s  = v.x + v.y + v.z + v.w;
  float sq = v.x * v.x + v.y * v.y + v.z * v.z + v.w * v.w;
#pragma unroll
  for (int m = 16; m >= 1; m >>= 1) {
    s  += __shfl_xor(s, m, 32);
    sq += __shfl_xor(sq, m, 32);
  }
  const float mean = s * (1.0f / DIM);
  const float var  = sq * (1.0f / DIM) - mean * mean;
  const float inv  = rsqrtf(var + 1e-8f);

  const float4 gg = ((const float4*)g)[lane];
  const float4 bb = ((const float4*)b)[lane];
  float4 o;
  o.x = (v.x - mean) * inv * gg.x + bb.x;
  o.y = (v.y - mean) * inv * gg.y + bb.y;
  o.z = (v.z - mean) * inv * gg.z + bb.z;
  o.w = (v.w - mean) * inv * gg.w + bb.w;
  ((float4*)(out + row * DIM))[lane] = o;
}

// ---------------------------------------------------------------------------
// Kernel 4: attention for one transformer block, residual fused.
// grid (L/16 qtiles, H). scores = q.k^T (WMMA, K=hd=32) + logtw, mask,
// softmax over 512 keys in LDS, weights -> d_out, out = w.v (WMMA, K=512),
// epilogue writes resid = Q + out.
// ---------------------------------------------------------------------------
#define WPAD (LSEQ + 4)   // stride pad: kills LDS bank conflicts on column reads

__global__ __launch_bounds__(256)
void attn_kernel(const float* __restrict__ q,       // [L, D]
                 const float* __restrict__ k,       // [L, D]
                 const float* __restrict__ v,       // [L, D]
                 const float* __restrict__ Qres,    // [L, D] residual input
                 const float* __restrict__ logtw,   // [L, L]
                 const float* __restrict__ keep,    // [L] 0/1
                 float* __restrict__ attn_w_out,    // [H, L, L]
                 float* __restrict__ resid_out)     // [L, D] = Qres + attn out
{
  const int tid   = threadIdx.x;
  const int lane  = tid & 31;
  const int wave  = tid >> 5;
  const int ln15  = lane & 15;
  const int halfk = (lane >> 4) * 8;
  const int qtile = blockIdx.x;
  const int h     = blockIdx.y;
  const int hoff  = h * HDIM;

  __shared__ float w_lds[16][WPAD];

  // q A-fragment (16 queries x K=32), reused across all key tiles
  v16h afrag;
  {
    const float* qp = q + (qtile * 16 + ln15) * DIM + hoff + halfk;
#pragma unroll
    for (int i = 0; i < 8; ++i) {
      afrag[i]     = (_Float16)qp[i];
      afrag[i + 8] = (_Float16)qp[16 + i];
    }
  }
  const float scale = 0.17677669529663687f;   // 1/sqrt(hd)

  // ---- stage 1: scores -> LDS (each wave does 4 key tiles) ----
  for (int kt = wave; kt < LSEQ / 16; kt += 8) {
    v16h bfrag;
    const float* kp = k + (kt * 16 + ln15) * DIM + hoff + halfk;
#pragma unroll
    for (int i = 0; i < 8; ++i) {
      bfrag[i]     = (_Float16)kp[i];
      bfrag[i + 8] = (_Float16)kp[16 + i];
    }
    v8f c = {};
    c = wmma16x16x32(afrag, bfrag, c);
#pragma unroll
    for (int r = 0; r < 8; ++r) {
      const int qr = qtile * 16 + halfk + r;
      const int kk = kt * 16 + ln15;
      float s = (c[r] + logtw[qr * LSEQ + kk]) * scale;
      const bool masked = (kk > qr) || (keep[qr] == 0.0f);
      w_lds[halfk + r][kk] = masked ? NEGV : s;
    }
  }
  __syncthreads();

  // ---- stage 2: softmax per row (16 threads per row) + write weights ----
  {
    const int m   = tid >> 4;
    const int s16 = tid & 15;
    float mx = -3.0e38f;
    for (int kk = s16; kk < LSEQ; kk += 16) mx = fmaxf(mx, w_lds[m][kk]);
#pragma unroll
    for (int mm = 8; mm >= 1; mm >>= 1) mx = fmaxf(mx, __shfl_xor(mx, mm, 32));
    float sum = 0.0f;
    for (int kk = s16; kk < LSEQ; kk += 16) {
      const float e = __expf(w_lds[m][kk] - mx);
      w_lds[m][kk] = e;
      sum += e;
    }
#pragma unroll
    for (int mm = 8; mm >= 1; mm >>= 1) sum += __shfl_xor(sum, mm, 32);
    const float inv = 1.0f / sum;
    const int qr = qtile * 16 + m;
    float* wout = attn_w_out + ((long)h * LSEQ + qr) * LSEQ;
    for (int kk = s16; kk < LSEQ; kk += 16) {
      const float wv = w_lds[m][kk] * inv;
      w_lds[m][kk] = wv;
      wout[kk] = wv;
    }
  }
  __syncthreads();

  // ---- stage 3: out = w @ v, residual-fused (waves 0..1 cover 32 cols) ----
  if (wave < 2) {
    const int col = hoff + wave * 16 + ln15;
    v8f c = {};
    for (int kc = 0; kc < LSEQ / 32; ++kc) {
      v16h a, bf;
#pragma unroll
      for (int i = 0; i < 8; ++i) {
        a[i]     = (_Float16)w_lds[ln15][kc * 32 + halfk + i];
        a[i + 8] = (_Float16)w_lds[ln15][kc * 32 + halfk + 16 + i];
      }
#pragma unroll
      for (int i = 0; i < 8; ++i) {
        bf[i]     = (_Float16)v[(kc * 32 + halfk + i) * DIM + col];
        bf[i + 8] = (_Float16)v[(kc * 32 + halfk + 16 + i) * DIM + col];
      }
      c = wmma16x16x32(a, bf, c);
    }
#pragma unroll
    for (int r = 0; r < 8; ++r) {
      const int row = qtile * 16 + halfk + r;
      resid_out[row * DIM + col] = Qres[row * DIM + col] + c[r];
    }
  }
}

// ---------------------------------------------------------------------------
// Elementwise helpers
// ---------------------------------------------------------------------------
__global__ void mask_seqs_kernel(const float* __restrict__ seqs,
                                 const int* __restrict__ log_seqs,
                                 float* __restrict__ out,
                                 float* __restrict__ keep)
{
  const int row = blockIdx.x;
  const float kv = (log_seqs[row] != 0) ? 1.0f : 0.0f;
  if (threadIdx.x == 0) keep[row] = kv;
  out[row * DIM + threadIdx.x] = seqs[row * DIM + threadIdx.x] * kv;
}

__global__ void ffn_final_kernel(const float* __restrict__ x,
                                 const float* __restrict__ y,
                                 const float* __restrict__ keep,
                                 float* __restrict__ out, int n)
{
  const int i = blockIdx.x * blockDim.x + threadIdx.x;
  if (i < n) out[i] = (x[i] + y[i]) * keep[i / DIM];
}

// ---------------------------------------------------------------------------
// Orchestration
// ---------------------------------------------------------------------------
extern "C" void kernel_launch(void* const* d_in, const int* in_sizes, int n_in,
                              void* d_out, int out_size, void* d_ws, size_t ws_size,
                              hipStream_t stream)
{
  (void)in_sizes; (void)n_in; (void)out_size; (void)ws_size;

  const int*   log_seqs = (const int*)  d_in[0];
  const float* seqs_in  = (const float*)d_in[1];
  const float* tmK      = (const float*)d_in[2];
  const float* Wq   = (const float*)d_in[3];
  const float* bq   = (const float*)d_in[4];
  const float* Wk   = (const float*)d_in[5];
  const float* bk   = (const float*)d_in[6];
  const float* Wv   = (const float*)d_in[7];
  const float* bv   = (const float*)d_in[8];
  const float* Wt   = (const float*)d_in[9];
  const float* bt   = (const float*)d_in[10];
  const float* Wtp  = (const float*)d_in[11];
  const float* btp  = (const float*)d_in[12];
  const float* ln1g = (const float*)d_in[13];
  const float* ln1b = (const float*)d_in[14];
  const float* ln2g = (const float*)d_in[15];
  const float* ln2b = (const float*)d_in[16];
  const float* W1   = (const float*)d_in[17];
  const float* b1   = (const float*)d_in[18];
  const float* W2   = (const float*)d_in[19];
  const float* b2   = (const float*)d_in[20];
  const float* lnfg = (const float*)d_in[21];
  const float* lnfb = (const float*)d_in[22];

  float* out = (float*)d_out;            // [L*D] log_feats, then NB x [H*L*L]

  const int N = LSEQ * DIM;              // 65536
  float* ws     = (float*)d_ws;
  float* logtw  = ws;                    // NB * L*L
  float* keep   = logtw + (long)NBLK * LSEQ * LSEQ;
  float* seqs   = keep + LSEQ;           // current hidden state
  float* bufQ   = seqs + N;
  float* bufq   = bufQ + N;
  float* bufk   = bufq + N;
  float* bufv   = bufk + N;
  float* bufX   = bufv + N;
  float* bufH   = bufX + N;
  float* bufY   = bufH + N;

  // Time-bias tables for both blocks (input-only -> done up front; this is
  // the 2 x 8.6 GFLOP / 134MB-stream hot loop)
  time_bias_kernel<<<dim3(1024, NBLK), 256, 0, stream>>>(
      tmK, Wt, bt, Wtp, btp, logtw);

  // seqs = seqs_in * keep
  mask_seqs_kernel<<<LSEQ, DIM, 0, stream>>>(seqs_in, log_seqs, seqs, keep);

  for (int i = 0; i < NBLK; ++i) {
    const long wo = (long)i * DIM * DIM;
    const long vo = (long)i * DIM;

    // Q = LN1(seqs)
    ln_kernel<<<LSEQ / 4, 128, 0, stream>>>(seqs, ln1g + vo, ln1b + vo, bufQ);
    // q/k/v projections
    gemm_tile16<<<LSEQ / 16, 256, 0, stream>>>(bufQ, Wq + wo, bq + vo, bufq, 0);
    gemm_tile16<<<LSEQ / 16, 256, 0, stream>>>(seqs, Wk + wo, bk + vo, bufk, 0);
    gemm_tile16<<<LSEQ / 16, 256, 0, stream>>>(seqs, Wv + wo, bv + vo, bufv, 0);

    // attention; weights stream straight into the output tuple region;
    // epilogue writes seqs = Q + attn_out (residual fused)
    float* attn_w = out + N + (long)i * NHEAD * LSEQ * LSEQ;
    attn_kernel<<<dim3(LSEQ / 16, NHEAD), 256, 0, stream>>>(
        bufq, bufk, bufv, bufQ, logtw + (long)i * LSEQ * LSEQ, keep,
        attn_w, seqs);

    // x = LN2(seqs); y = relu(x@W1+b1)@W2+b2; seqs = (x+y)*keep
    ln_kernel<<<LSEQ / 4, 128, 0, stream>>>(seqs, ln2g + vo, ln2b + vo, bufX);
    gemm_tile16<<<LSEQ / 16, 256, 0, stream>>>(bufX, W1 + wo, b1 + vo, bufH, 1);
    gemm_tile16<<<LSEQ / 16, 256, 0, stream>>>(bufH, W2 + wo, b2 + vo, bufY, 0);
    ffn_final_kernel<<<N / 256, 256, 0, stream>>>(bufX, bufY, keep, seqs, N);
  }

  // log_feats = LNf(seqs)
  ln_kernel<<<LSEQ / 4, 128, 0, stream>>>(seqs, lnfg, lnfb, out);
}
